// conformer_9560597200918
// MI455X (gfx1250) — matrix operand order repaired and checked by reading, not compiled, not run on hardware
//
#include <hip/hip_runtime.h>
#include <cstdint>
#include <cstddef>

// ---------------------------------------------------------------------------
// Types / helpers
// ---------------------------------------------------------------------------
typedef __bf16 bf16_t;
typedef __attribute__((ext_vector_type(16))) __bf16 v16bf;
typedef __attribute__((ext_vector_type(8)))  __bf16 v8bf;
typedef __attribute__((ext_vector_type(8)))  float  v8f;

#define BT   4096   // B*T
#define DMODEL 512
#define TSEQ 1024
#define NHEAD 8
#define HDIM 64

__device__ __forceinline__ bf16_t f2bf(float f) {
    uint32_t u = __builtin_bit_cast(uint32_t, f);
    uint32_t r = u + 0x7FFFu + ((u >> 16) & 1u);   // round-to-nearest-even
    uint16_t s = (uint16_t)(r >> 16);
    return __builtin_bit_cast(bf16_t, s);
}
__device__ __forceinline__ float bf2f(bf16_t h) {
    uint16_t s = __builtin_bit_cast(uint16_t, h);
    uint32_t u = ((uint32_t)s) << 16;
    return __builtin_bit_cast(float, u);
}
__device__ __forceinline__ v8f zero8() {
    v8f z;
#pragma unroll
    for (int i = 0; i < 8; ++i) z[i] = 0.0f;
    return z;
}
__device__ __forceinline__ v8f wmma_bf16(v16bf a, v16bf b, v8f c) {
    return __builtin_amdgcn_wmma_f32_16x16x32_bf16(false, a, false, b, (short)0, c,
                                                   false, false);
}

// A tile 16x32 (MxK), bf16, row-major source (stride lda elements).
// Layout used here: lane m=lane&15, half=lane>>4; elements 0..15 hold
// K=16h+0..15 (contiguous, matches load_b_nk).
__device__ __forceinline__ v16bf load_a16x32(const bf16_t* base, int lda) {
    int lane = threadIdx.x & 31;
    int m = lane & 15, hf = lane >> 4;
    const bf16_t* p = base + (size_t)m * lda + hf * 16;
    v16bf r = *(const v16bf*)p;          // K = 16h .. 16h+15
    return r;
}

// B tile 32x16 (KxN) from a [N,K] row-major source (stride ldb elements).
// Lane n=lane&15 is column N; half=lane>>4 picks K range 16h..16h+15
// (one contiguous 32-byte load per lane).
__device__ __forceinline__ v16bf load_b_nk(const bf16_t* base, int ldb) {
    int lane = threadIdx.x & 31;
    int n = lane & 15, hf = lane >> 4;
    return *(const v16bf*)(base + (size_t)n * ldb + hf * 16);
}

// ---------------------------------------------------------------------------
// Weight conversion kernels (f32 -> bf16, optionally transposed to [N,K])
// ---------------------------------------------------------------------------
__global__ void __launch_bounds__(256) convert_wt(const float* __restrict__ src,
                                                  bf16_t* __restrict__ dst,
                                                  int K, int N) {
    int idx = blockIdx.x * 256 + threadIdx.x;
    if (idx >= N * K) return;
    int n = idx / K, k = idx % K;
    dst[idx] = f2bf(src[(size_t)k * N + n]);
}
__global__ void __launch_bounds__(256) convert_plain(const float* __restrict__ src,
                                                     bf16_t* __restrict__ dst,
                                                     int count) {
    int idx = blockIdx.x * 256 + threadIdx.x;
    if (idx < count) dst[idx] = f2bf(src[idx]);
}

// ---------------------------------------------------------------------------
// LayerNorm: one row (D=512) per 256-thread block. bf16 or f32 output.
// ---------------------------------------------------------------------------
__global__ void __launch_bounds__(256) ln_kernel(const float* __restrict__ x,
                                                 const float* __restrict__ g,
                                                 const float* __restrict__ b,
                                                 bf16_t* __restrict__ outb,
                                                 float* __restrict__ outf) {
    __shared__ float red[2][8];
    int row = blockIdx.x, tid = threadIdx.x;
    int lane = tid & 31, wid = tid >> 5;
    const float* xr = x + (size_t)row * DMODEL;
    float v0 = xr[tid], v1 = xr[tid + 256];
    float s = v0 + v1, q = v0 * v0 + v1 * v1;
#pragma unroll
    for (int off = 16; off >= 1; off >>= 1) {
        s += __shfl_xor(s, off, 32);
        q += __shfl_xor(q, off, 32);
    }
    if (lane == 0) { red[0][wid] = s; red[1][wid] = q; }
    __syncthreads();
    float ts = 0.0f, tq = 0.0f;
#pragma unroll
    for (int i = 0; i < 8; ++i) { ts += red[0][i]; tq += red[1][i]; }
    float mean = ts * (1.0f / DMODEL);
    float var  = tq * (1.0f / DMODEL) - mean * mean;
    float rstd = rsqrtf(var + 1e-5f);
    float y0 = (v0 - mean) * rstd * g[tid] + b[tid];
    float y1 = (v1 - mean) * rstd * g[tid + 256] + b[tid + 256];
    if (outf) {
        outf[(size_t)row * DMODEL + tid]       = y0;
        outf[(size_t)row * DMODEL + tid + 256] = y1;
    } else {
        outb[(size_t)row * DMODEL + tid]       = f2bf(y0);
        outb[(size_t)row * DMODEL + tid + 256] = f2bf(y1);
    }
}

// ---------------------------------------------------------------------------
// Generic WMMA GEMM: C[M,N] = A[M,K](bf16) * Bt[N,K](bf16), fused epilogue.
// Wave computes a 16x64 tile. 8 waves / block.
// EPI: 0 = bf16 (acc+bias)*scale; 1 = bf16 swish(acc+bias);
//      2 = f32 acc+bias+res;      3 = f32 0.5*(acc+bias)+res
// ---------------------------------------------------------------------------
template <int EPI>
__global__ void __launch_bounds__(256) gemm_ws(const bf16_t* __restrict__ A,
                                               const bf16_t* __restrict__ Bt,
                                               const float* __restrict__ bias,
                                               const float* __restrict__ res,
                                               bf16_t* __restrict__ outb,
                                               float* __restrict__ outf,
                                               int M, int N, int K, float scale) {
    int wid  = threadIdx.x >> 5;
    int w    = blockIdx.x * 8 + wid;
    int mt   = M >> 4;
    int tM   = w % mt, tN = w / mt;
    int lane = threadIdx.x & 31;
    int n = lane & 15, hf = lane >> 4;

    const bf16_t* Abase = A + (size_t)tM * 16 * K;
    v8f acc[4];
#pragma unroll
    for (int t = 0; t < 4; ++t) acc[t] = zero8();

    for (int k0 = 0; k0 < K; k0 += 32) {
        __builtin_prefetch(Abase + k0 + 128, 0, 1);   // global_prefetch_b8
        v16bf a = load_a16x32(Abase + k0, K);
#pragma unroll
        for (int nt = 0; nt < 4; ++nt) {
            v16bf b = load_b_nk(Bt + (size_t)(tN * 64 + nt * 16) * K + k0, K);
            acc[nt] = wmma_bf16(a, b, acc[nt]);
        }
    }

#pragma unroll
    for (int nt = 0; nt < 4; ++nt) {
#pragma unroll
        for (int e = 0; e < 8; ++e) {
            int row = tM * 16 + e + 8 * hf;
            int col = tN * 64 + nt * 16 + n;
            size_t idx = (size_t)row * N + col;
            float v = acc[nt][e] + (bias ? bias[col] : 0.0f);
            v *= scale;
            if constexpr (EPI == 0) {
                outb[idx] = f2bf(v);
            } else if constexpr (EPI == 1) {
                float sg = 1.0f / (1.0f + __expf(-v));
                outb[idx] = f2bf(v * sg);
            } else if constexpr (EPI == 2) {
                outf[idx] = v + res[idx];
            } else {
                outf[idx] = 0.5f * v + res[idx];
            }
        }
    }
}

// ---------------------------------------------------------------------------
// Attention: flash-style, wave = (b, h, 16 query rows), streaming 32-key
// chunks. q pre-scaled by 1/8. Rel-pos bias via local q@rel^T WMMA (16x48
// window) gathered through LDS with the skew j = ti - si + 31.
// ---------------------------------------------------------------------------
__global__ void __launch_bounds__(256) attn_kernel(const bf16_t* __restrict__ Q,
                                                   const bf16_t* __restrict__ KV,
                                                   const bf16_t* __restrict__ REL,
                                                   bf16_t* __restrict__ CTX) {
    __shared__ float  s_qr[8 * 16 * 48];   // per-wave 16x48 q@rel^T tile
    __shared__ bf16_t s_p [8 * 16 * 32];   // per-wave 16x32 probs (bf16)

    int wid  = threadIdx.x >> 5;
    int lane = threadIdx.x & 31;
    int n = lane & 15, hf = lane >> 4;
    int w  = blockIdx.x * 8 + wid;
    int tb = w & 63;             // T/16 = 64 tiles
    int bh = w >> 6;             // 0..31
    int h  = bh & 7, b = bh >> 3;
    int t0 = tb * 16;

    float*  qr = s_qr + wid * (16 * 48);
    bf16_t* pb = s_p  + wid * (16 * 32);

    const bf16_t* qbase = Q + ((size_t)(b * TSEQ + t0)) * DMODEL + h * HDIM;
    v16bf qa0 = load_a16x32(qbase,      DMODEL);   // K = d 0..31
    v16bf qa1 = load_a16x32(qbase + 32, DMODEL);   // K = d 32..63

    float mrow[8], lrow[8];
#pragma unroll
    for (int e = 0; e < 8; ++e) { mrow[e] = -3.0e38f; lrow[e] = 0.0f; }
    v8f ctx[4];
#pragma unroll
    for (int t = 0; t < 4; ++t) ctx[t] = zero8();

    for (int s0 = 0; s0 < TSEQ; s0 += 32) {
        // --- scores = q @ k^T (two 16-col subtiles) ---
        const bf16_t* kbase = KV + ((size_t)(b * TSEQ + s0)) * 1024 + h * HDIM;
        v8f sa0 = zero8(), sa1 = zero8();
        sa0 = wmma_bf16(qa0, load_b_nk(kbase,                 1024), sa0);
        sa0 = wmma_bf16(qa1, load_b_nk(kbase + 32,            1024), sa0);
        sa1 = wmma_bf16(qa0, load_b_nk(kbase + 16 * 1024,      1024), sa1);
        sa1 = wmma_bf16(qa1, load_b_nk(kbase + 16 * 1024 + 32, 1024), sa1);

        // --- rel-pos: qr[ti, j] = q[ti] . rel_pemb[clip(base-31+j)] ---
        int base = t0 - s0;
#pragma unroll
        for (int jt = 0; jt < 3; ++jt) {
            int j  = jt * 16 + n;
            int rp = base - 31 + j;
            rp = rp < -512 ? -512 : (rp > 512 ? 512 : rp);
            rp += 512;
            const bf16_t* rrow = REL + (size_t)rp * HDIM;
            v16bf rb0 = *(const v16bf*)(rrow + hf * 16);
            v16bf rb1 = *(const v16bf*)(rrow + 32 + hf * 16);
            v8f qacc = zero8();
            qacc = wmma_bf16(qa0, rb0, qacc);
            qacc = wmma_bf16(qa1, rb1, qacc);
#pragma unroll
            for (int e = 0; e < 8; ++e)
                qr[(e + 8 * hf) * 48 + jt * 16 + n] = qacc[e];
        }
        asm volatile("s_wait_dscnt 0" ::: "memory");

        // --- bias add + online softmax update ---
        float fac[8];
#pragma unroll
        for (int e = 0; e < 8; ++e) {
            int ti = e + 8 * hf;
            float v0 = sa0[e] + qr[ti * 48 + (ti - n + 31)];        // si = n
            float v1 = sa1[e] + qr[ti * 48 + (ti - n + 15)];        // si = 16+n
            float rmax = fmaxf(v0, v1);
#pragma unroll
            for (int off = 1; off < 16; off <<= 1)
                rmax = fmaxf(rmax, __shfl_xor(rmax, off, 32));
            float mnew = fmaxf(mrow[e], rmax);
            float f  = __expf(mrow[e] - mnew);
            float p0 = __expf(v0 - mnew);
            float p1 = __expf(v1 - mnew);
            float rs = p0 + p1;
#pragma unroll
            for (int off = 1; off < 16; off <<= 1)
                rs += __shfl_xor(rs, off, 32);
            lrow[e] = lrow[e] * f + rs;
            mrow[e] = mnew;
            fac[e]  = f;
            pb[ti * 32 + n]      = f2bf(p0);
            pb[ti * 32 + 16 + n] = f2bf(p1);
        }
        asm volatile("s_wait_dscnt 0" ::: "memory");

#pragma unroll
        for (int nt = 0; nt < 4; ++nt)
#pragma unroll
            for (int e = 0; e < 8; ++e) ctx[nt][e] *= fac[e];

        // --- ctx += P @ V ---
        v16bf pa = load_a16x32(pb, 32);   // from LDS
        const bf16_t* vbase =
            KV + ((size_t)(b * TSEQ + s0)) * 1024 + 512 + h * HDIM;
#pragma unroll
        for (int nt = 0; nt < 4; ++nt) {
            v16bf bv;
#pragma unroll
            for (int vv = 0; vv < 8; ++vv) {
                int kk = hf * 16 + 2 * vv;
                const bf16_t* p = vbase + (size_t)kk * 1024 + nt * 16 + n;
                bv[2 * vv]     = p[0];
                bv[2 * vv + 1] = p[1024];
            }
            ctx[nt] = wmma_bf16(pa, bv, ctx[nt]);
        }
    }

    // --- normalize and store ctx (bf16, [BT, 512] with head offset) ---
#pragma unroll
    for (int nt = 0; nt < 4; ++nt) {
#pragma unroll
        for (int e = 0; e < 8; ++e) {
            int ti = e + 8 * hf;
            float v = ctx[nt][e] / lrow[e];
            CTX[((size_t)(b * TSEQ + t0 + ti)) * DMODEL + h * HDIM + nt * 16 + n] =
                f2bf(v);
        }
    }
}

// ---------------------------------------------------------------------------
// GLU: c = a * sigmoid(g), in [BT,2048] -> out [BT,1024]
// ---------------------------------------------------------------------------
__global__ void __launch_bounds__(256) glu_kernel(const bf16_t* __restrict__ in,
                                                  bf16_t* __restrict__ out) {
    int gid = blockIdx.x * 256 + threadIdx.x;
    if (gid >= BT * 1024) return;
    int r = gid >> 10, c = gid & 1023;
    float a = bf2f(in[(size_t)r * 2048 + c]);
    float g = bf2f(in[(size_t)r * 2048 + 1024 + c]);
    out[gid] = f2bf(a * (1.0f / (1.0f + __expf(-g))));
}

// ---------------------------------------------------------------------------
// Depthwise conv (K=31, SAME) + BN(inference) + swish. [B,T,1024]
// ---------------------------------------------------------------------------
__global__ void __launch_bounds__(256) dwconv_kernel(
    const bf16_t* __restrict__ in, const float* __restrict__ w,
    const float* __restrict__ cb, const float* __restrict__ bg,
    const float* __restrict__ bb, const float* __restrict__ bmean,
    const float* __restrict__ bvar, bf16_t* __restrict__ out) {
    int gid = blockIdx.x * 256 + threadIdx.x;
    if (gid >= BT * 1024) return;
    int c = gid & 1023;
    int r = gid >> 10;
    int t = r & (TSEQ - 1);
    int b = r >> 10;
    float acc = 0.0f;
#pragma unroll
    for (int k = 0; k < 31; ++k) {
        int tt = t + k - 15;
        if ((unsigned)tt < (unsigned)TSEQ)
            acc += bf2f(in[((size_t)(b * TSEQ + tt) << 10) + c]) * w[k * 1024 + c];
    }
    float y = acc + cb[c];
    y = (y - bmean[c]) * rsqrtf(bvar[c] + 1e-5f) * bg[c] + bb[c];
    out[gid] = f2bf(y * (1.0f / (1.0f + __expf(-y))));
}

// ---------------------------------------------------------------------------
// Host launcher
// ---------------------------------------------------------------------------
extern "C" void kernel_launch(void* const* d_in, const int* in_sizes, int n_in,
                              void* d_out, int out_size, void* d_ws, size_t ws_size,
                              hipStream_t stream) {
    (void)in_sizes; (void)n_in; (void)out_size; (void)ws_size;
    const float* x       = (const float*)d_in[0];
    const float* ff1_ng  = (const float*)d_in[1];
    const float* ff1_nb  = (const float*)d_in[2];
    const float* ff1_w1  = (const float*)d_in[3];
    const float* ff1_b1  = (const float*)d_in[4];
    const float* ff1_w2  = (const float*)d_in[5];
    const float* ff1_b2  = (const float*)d_in[6];
    const float* attn_ng = (const float*)d_in[7];
    const float* attn_nb = (const float*)d_in[8];
    const float* q_w     = (const float*)d_in[9];
    const float* kv_w    = (const float*)d_in[10];
    const float* out_w   = (const float*)d_in[11];
    const float* out_b   = (const float*)d_in[12];
    const float* rel_pemb= (const float*)d_in[13];
    const float* conv_ng = (const float*)d_in[14];
    const float* conv_nb = (const float*)d_in[15];
    const float* conv1_w = (const float*)d_in[16];
    const float* conv1_b = (const float*)d_in[17];
    const float* dconv_w = (const float*)d_in[18];
    const float* dconv_b = (const float*)d_in[19];
    const float* bn_g    = (const float*)d_in[20];
    const float* bn_b    = (const float*)d_in[21];
    const float* bn_mean = (const float*)d_in[22];
    const float* bn_var  = (const float*)d_in[23];
    const float* conv2_w = (const float*)d_in[24];
    const float* conv2_b = (const float*)d_in[25];
    const float* ff2_ng  = (const float*)d_in[26];
    const float* ff2_nb  = (const float*)d_in[27];
    const float* ff2_w1  = (const float*)d_in[28];
    const float* ff2_b1  = (const float*)d_in[29];
    const float* ff2_w2  = (const float*)d_in[30];
    const float* ff2_b2  = (const float*)d_in[31];
    const float* norm_g  = (const float*)d_in[32];
    const float* norm_b  = (const float*)d_in[33];

    uint8_t* ws = (uint8_t*)d_ws;
    size_t off = 0;
    auto alloc = [&](size_t bytes) -> void* {
        void* p = ws + off;
        off += (bytes + 255) & ~(size_t)255;
        return p;
    };
    float*  X     = (float*) alloc((size_t)BT * 512 * 4);
    bf16_t* XA    = (bf16_t*)alloc((size_t)BT * 512 * 2);
    bf16_t* Hbuf  = (bf16_t*)alloc((size_t)BT * 2048 * 2);
    bf16_t* GLU   = (bf16_t*)alloc((size_t)BT * 1024 * 2);
    bf16_t* DC    = (bf16_t*)alloc((size_t)BT * 1024 * 2);
    bf16_t* Qb    = (bf16_t*)alloc((size_t)BT * 512 * 2);
    bf16_t* KVb   = (bf16_t*)alloc((size_t)BT * 1024 * 2);
    bf16_t* CTXb  = (bf16_t*)alloc((size_t)BT * 512 * 2);
    bf16_t* Wff1a = (bf16_t*)alloc((size_t)2048 * 512 * 2);
    bf16_t* Wff1b = (bf16_t*)alloc((size_t)512 * 2048 * 2);
    bf16_t* Wq    = (bf16_t*)alloc((size_t)512 * 512 * 2);
    bf16_t* Wkv   = (bf16_t*)alloc((size_t)1024 * 512 * 2);
    bf16_t* Wout  = (bf16_t*)alloc((size_t)512 * 512 * 2);
    bf16_t* Wc1   = (bf16_t*)alloc((size_t)2048 * 512 * 2);
    bf16_t* Wc2   = (bf16_t*)alloc((size_t)512 * 1024 * 2);
    bf16_t* Wff2a = (bf16_t*)alloc((size_t)2048 * 512 * 2);
    bf16_t* Wff2b = (bf16_t*)alloc((size_t)512 * 2048 * 2);
    bf16_t* RELb  = (bf16_t*)alloc((size_t)1025 * 64 * 2);

    auto blk = [](size_t n) { return dim3((unsigned)((n + 255) / 256)); };

    // Weight conversion (f32 [K,N] -> bf16 [N,K])
    convert_wt<<<blk(2048 * 512), 256, 0, stream>>>(ff1_w1,  Wff1a, 512, 2048);
    convert_wt<<<blk(512 * 2048), 256, 0, stream>>>(ff1_w2,  Wff1b, 2048, 512);
    convert_wt<<<blk(512 * 512),  256, 0, stream>>>(q_w,     Wq,    512, 512);
    convert_wt<<<blk(1024 * 512), 256, 0, stream>>>(kv_w,    Wkv,   512, 1024);
    convert_wt<<<blk(512 * 512),  256, 0, stream>>>(out_w,   Wout,  512, 512);
    convert_wt<<<blk(2048 * 512), 256, 0, stream>>>(conv1_w, Wc1,   512, 2048);
    convert_wt<<<blk(512 * 1024), 256, 0, stream>>>(conv2_w, Wc2,   1024, 512);
    convert_wt<<<blk(2048 * 512), 256, 0, stream>>>(ff2_w1,  Wff2a, 512, 2048);
    convert_wt<<<blk(512 * 2048), 256, 0, stream>>>(ff2_w2,  Wff2b, 2048, 512);
    convert_plain<<<blk(1025 * 64), 256, 0, stream>>>(rel_pemb, RELb, 1025 * 64);

    // FF1 (half-step residual)
    ln_kernel<<<BT, 256, 0, stream>>>(x, ff1_ng, ff1_nb, XA, nullptr);
    gemm_ws<1><<<1024, 256, 0, stream>>>(XA, Wff1a, ff1_b1, nullptr, Hbuf, nullptr,
                                         BT, 2048, 512, 1.0f);
    gemm_ws<3><<<256, 256, 0, stream>>>(Hbuf, Wff1b, ff1_b2, x, nullptr, X,
                                        BT, 512, 2048, 1.0f);

    // Attention
    ln_kernel<<<BT, 256, 0, stream>>>(X, attn_ng, attn_nb, XA, nullptr);
    gemm_ws<0><<<256, 256, 0, stream>>>(XA, Wq, nullptr, nullptr, Qb, nullptr,
                                        BT, 512, 512, 0.125f);   // scale = HD^-0.5
    gemm_ws<0><<<512, 256, 0, stream>>>(XA, Wkv, nullptr, nullptr, KVb, nullptr,
                                        BT, 1024, 512, 1.0f);
    attn_kernel<<<256, 256, 0, stream>>>(Qb, KVb, RELb, CTXb);
    gemm_ws<2><<<256, 256, 0, stream>>>(CTXb, Wout, out_b, X, nullptr, X,
                                        BT, 512, 512, 1.0f);

    // Conv module
    ln_kernel<<<BT, 256, 0, stream>>>(X, conv_ng, conv_nb, XA, nullptr);
    gemm_ws<0><<<1024, 256, 0, stream>>>(XA, Wc1, conv1_b, nullptr, Hbuf, nullptr,
                                         BT, 2048, 512, 1.0f);
    glu_kernel<<<blk((size_t)BT * 1024), 256, 0, stream>>>(Hbuf, GLU);
    dwconv_kernel<<<blk((size_t)BT * 1024), 256, 0, stream>>>(
        GLU, dconv_w, dconv_b, bn_g, bn_b, bn_mean, bn_var, DC);
    gemm_ws<2><<<256, 256, 0, stream>>>(DC, Wc2, conv2_b, X, nullptr, X,
                                        BT, 512, 1024, 1.0f);

    // FF2 (half-step residual) + final norm
    ln_kernel<<<BT, 256, 0, stream>>>(X, ff2_ng, ff2_nb, XA, nullptr);
    gemm_ws<1><<<1024, 256, 0, stream>>>(XA, Wff2a, ff2_b1, nullptr, Hbuf, nullptr,
                                         BT, 2048, 512, 1.0f);
    gemm_ws<3><<<256, 256, 0, stream>>>(Hbuf, Wff2b, ff2_b2, X, nullptr, X,
                                        BT, 512, 2048, 1.0f);
    ln_kernel<<<BT, 256, 0, stream>>>(X, norm_g, norm_b, nullptr, (float*)d_out);
}
